// ConversationLSTMStack_47828755808363
// MI455X (gfx1250) — matrix operand order repaired and checked by reading, not compiled, not run on hardware
//
#include <hip/hip_runtime.h>
#include <hip/hip_bf16.h>
#include <math.h>

// ---------------------------------------------------------------------------
// Problem constants (from the reference): B=64, S=2048, D=H=768, L=2
// ---------------------------------------------------------------------------
constexpr int  BB    = 64;
constexpr int  SS    = 2048;
constexpr int  HH    = 768;          // D == H == 768
constexpr int  LL    = 2;
constexpr int  FOURH = 4 * HH;       // 3072
constexpr long long MROWS = (long long)BB * SS;  // 131072

typedef unsigned short u16;
typedef __attribute__((ext_vector_type(16))) __bf16 v16bf;
typedef __attribute__((ext_vector_type(8)))  float  v8f;

union FragBF {            // 16 bf16 = 32 bytes = two 16-byte chunks
    v16bf v;
    uint4 q[2];
};

__device__ __forceinline__ u16 f32_to_bf16(float f) {
    union { float f; unsigned int u; } v;
    v.f = f;
    unsigned int r = v.u + 0x7FFFu + ((v.u >> 16) & 1u);   // round-to-nearest-even
    return (u16)(r >> 16);
}

__device__ __forceinline__ float sigmoidf_(float x) {
    return 1.0f / (1.0f + __expf(-x));
}

// A and B fragment chunks both sit 16 halves apart.
__device__ __forceinline__ void lda(FragBF& f, const u16* p) {
    f.q[0] = *reinterpret_cast<const uint4*>(p);
    f.q[1] = *reinterpret_cast<const uint4*>(p + 16);
}
__device__ __forceinline__ void ldb(FragBF& f, const u16* p) {
    f.q[0] = *reinterpret_cast<const uint4*>(p);
    f.q[1] = *reinterpret_cast<const uint4*>(p + 16);
}
__device__ __forceinline__ v8f wmma_(const FragBF& a, const FragBF& b, v8f c) {
    return __builtin_amdgcn_wmma_f32_16x16x32_bf16(
        false, a.v, false, b.v, (short)0, c, false, false);
}

// ---------------------------------------------------------------------------
// fp32 -> bf16 conversion (grid-stride)
// ---------------------------------------------------------------------------
__global__ void cvt_f32_bf16(const float* __restrict__ in,
                             u16* __restrict__ out, long long n) {
    long long i = (long long)blockIdx.x * blockDim.x + threadIdx.x;
    long long stride = (long long)gridDim.x * blockDim.x;
    for (; i < n; i += stride) out[i] = f32_to_bf16(in[i]);
}

// ---------------------------------------------------------------------------
// bf16 WMMA GEMM:  C[m][n] = sum_k A[m][k] * Bm[n][k]    (Bm is W, row-major [N,K])
// Workgroup: 256 threads = 8 waves arranged 2(M) x 4(N); each wave: 16M x 128N.
// K loop in steps of 32 using v_wmma_f32_16x16x32_bf16.
//
// Per K-step: issue ALL 18 vmem reads (1 A pair + 8 B pairs), then the 8
// WMMAs. sched_group_barrier pins this order so the scheduler cannot sink
// loads to their uses (which lets the RA collapse the fragment buffers and
// serialize load->wait0->wmma). With loads-then-WMMAs forced, the waitcnt
// pass emits staged partial waits: WMMA j waits only for its own pair while
// the remaining loads stay in flight (deep per-wave MLP against L2).
// ---------------------------------------------------------------------------
__global__ __launch_bounds__(256, 1)
void gemm_bf16_wmma(const u16* __restrict__ A,    // [M, K] bf16, row-major
                    const u16* __restrict__ Bm,   // [N, K] bf16, row-major (== W)
                    float* __restrict__ C,        // [M, N] fp32
                    int Kdim, int Ndim) {
    const int lane  = threadIdx.x & 31;
    const int wave  = threadIdx.x >> 5;   // 0..7
    const int wm    = wave & 1;           // 2 waves across M
    const int wn    = wave >> 1;          // 4 waves across N
    const int m0    = blockIdx.x * 32 + wm * 16;
    const int n0    = blockIdx.y * 512 + wn * 128;
    const int laneM = lane & 15;
    const int hi    = lane >> 4;          // 0 or 1

    v8f acc[8];
#pragma unroll
    for (int j = 0; j < 8; ++j)
#pragma unroll
        for (int r = 0; r < 8; ++r) acc[j][r] = 0.0f;

    // A fragment (16x32): lanes 0-15 hold K = k0+[0..7] & k0+[16..23];
    // lanes 16-31 hold K = k0+[8..15] & k0+[24..31].
    const u16* arow = A + (size_t)(m0 + laneM) * Kdim + hi * 8;
    // B fragment (32x16): lane's column = lane&15; lanes 0-15 hold
    // K = k0+[0..7] & k0+[16..23], lanes 16-31 hold K = k0+[8..15] & k0+[24..31].
    // Column n of B is row n of W (row-major [N,K]).
    const u16* brow[8];
#pragma unroll
    for (int j = 0; j < 8; ++j)
        brow[j] = Bm + (size_t)(n0 + j * 16 + laneM) * Kdim + hi * 8;

    for (int k0 = 0; k0 < Kdim; k0 += 32) {
        FragBF a, b[8];
        lda(a, arow + k0);
#pragma unroll
        for (int j = 0; j < 8; ++j)
            ldb(b[j], brow[j] + k0);

#pragma unroll
        for (int j = 0; j < 8; ++j)
            acc[j] = wmma_(a, b[j], acc[j]);

        // Scheduling contract for this K-step:
        //   group 1: 18 VMEM reads (mask 0x020)
        //   group 2:  8 WMMA       (mask 0x008, MFMA/WMMA class)
        __builtin_amdgcn_sched_group_barrier(0x020, 18, 0);
        __builtin_amdgcn_sched_group_barrier(0x008, 8, 0);
    }

    // C/D layout: lane l, VGPR r -> M = r + 8*(l>=16), N = l&15
#pragma unroll
    for (int j = 0; j < 8; ++j) {
        const int col = n0 + j * 16 + laneM;
        float* crow = C + (size_t)(m0 + hi * 8) * Ndim + col;
#pragma unroll
        for (int r = 0; r < 8; ++r)
            crow[(size_t)r * Ndim] = acc[j][r];
    }
}

// ---------------------------------------------------------------------------
// Layer-0 scan: per-(b,h) thread carries c across t; speaker change resets c.
// Writes h0 as bf16 (GEMM1 input) + final h/c for layer 0.
// gates rows are m = b*S + t (matching x [B,S,D] flattening).
// ---------------------------------------------------------------------------
__global__ __launch_bounds__(256)
void lstm_scan_layer0(const float* __restrict__ gates,       // [B*S, 4H]
                      const float* __restrict__ b_ih,        // [4H] (layer 0)
                      const float* __restrict__ b_hh,        // [4H] (layer 0)
                      const int* __restrict__ participants,  // [S]
                      u16* __restrict__ h0_bf,               // [B*S, H]
                      float* __restrict__ hfin,              // [B,H]
                      float* __restrict__ cfin) {            // [B,H]
    const int tid = blockIdx.x * blockDim.x + threadIdx.x;   // b*H + h
    const int h = tid % HH;
    const int b = tid / HH;

    const float bi = b_ih[0 * HH + h] + b_hh[0 * HH + h];
    const float bf = b_ih[1 * HH + h] + b_hh[1 * HH + h];
    const float bg = b_ih[2 * HH + h] + b_hh[2 * HH + h];
    const float bo = b_ih[3 * HH + h] + b_hh[3 * HH + h];

    const float* grow = gates + (size_t)b * SS * FOURH;

    float c = 0.0f, hv = 0.0f;
    int prev = 0;
    for (int t = 0; t < SS; ++t) {
        const int p = participants[t];
        if (t == 0 || p != prev) c = 0.0f;   // speaker-change reset (layer 0 only)
        prev = p;

        const float* g4 = grow + (size_t)t * FOURH;
        const float ig = sigmoidf_(g4[0 * HH + h] + bi);
        const float fg = sigmoidf_(g4[1 * HH + h] + bf);
        const float gg = tanhf    (g4[2 * HH + h] + bg);
        const float og = sigmoidf_(g4[3 * HH + h] + bo);

        c  = fg * c + ig * gg;
        hv = og * tanhf(c);
        h0_bf[((size_t)b * SS + t) * HH + h] = f32_to_bf16(hv);
    }
    hfin[(size_t)b * HH + h] = hv;
    cfin[(size_t)b * HH + h] = c;
}

// ---------------------------------------------------------------------------
// Layer-1 scan: no reset; writes outputs [S,B,H] fp32 + final h/c for layer 1.
// ---------------------------------------------------------------------------
__global__ __launch_bounds__(256)
void lstm_scan_layer1(const float* __restrict__ gates,  // [B*S, 4H]
                      const float* __restrict__ b_ih,   // [4H] (layer 1)
                      const float* __restrict__ b_hh,   // [4H] (layer 1)
                      float* __restrict__ outputs,      // [S,B,H]
                      float* __restrict__ hfin,         // [B,H]
                      float* __restrict__ cfin) {       // [B,H]
    const int tid = blockIdx.x * blockDim.x + threadIdx.x;   // b*H + h
    const int h = tid % HH;
    const int b = tid / HH;

    const float bi = b_ih[0 * HH + h] + b_hh[0 * HH + h];
    const float bf = b_ih[1 * HH + h] + b_hh[1 * HH + h];
    const float bg = b_ih[2 * HH + h] + b_hh[2 * HH + h];
    const float bo = b_ih[3 * HH + h] + b_hh[3 * HH + h];

    const float* grow = gates + (size_t)b * SS * FOURH;

    float c = 0.0f, hv = 0.0f;
    for (int t = 0; t < SS; ++t) {
        const float* g4 = grow + (size_t)t * FOURH;
        const float ig = sigmoidf_(g4[0 * HH + h] + bi);
        const float fg = sigmoidf_(g4[1 * HH + h] + bf);
        const float gg = tanhf    (g4[2 * HH + h] + bg);
        const float og = sigmoidf_(g4[3 * HH + h] + bo);

        c  = fg * c + ig * gg;
        hv = og * tanhf(c);
        outputs[((size_t)t * BB + b) * HH + h] = hv;
    }
    hfin[(size_t)b * HH + h] = hv;
    cfin[(size_t)b * HH + h] = c;
}

// ---------------------------------------------------------------------------
// Orchestration
//   d_in: 0:x [B,S,D]  1:W_ih [L,4H,D]  2:W_hh (unused)  3:b_ih [L,4H]
//         4:b_hh [L,4H]  5:participants [S]
//   d_out: outputs [S,B,H] ++ h_fin [L,B,H] ++ c_fin [L,B,H]   (fp32)
//   Workspace: Xb bf16 | Hb bf16 | Wb bf16 (both layers) | gates fp32 (reused)
// ---------------------------------------------------------------------------
extern "C" void kernel_launch(void* const* d_in, const int* /*in_sizes*/, int /*n_in*/,
                              void* d_out, int /*out_size*/, void* d_ws, size_t /*ws_size*/,
                              hipStream_t stream) {
    const float* x     = (const float*)d_in[0];
    const float* W_ih  = (const float*)d_in[1];
    const float* b_ih  = (const float*)d_in[3];
    const float* b_hh  = (const float*)d_in[4];
    const int*   parts = (const int*)d_in[5];

    float* out  = (float*)d_out;
    float* hfin = out + (size_t)SS * BB * HH;            // [L,B,H]
    float* cfin = hfin + (size_t)LL * BB * HH;           // [L,B,H]

    // Workspace carve-out (all chunks 16-byte aligned)
    u16*   Xb    = (u16*)d_ws;                           // [M, 768] bf16
    u16*   Hb    = Xb + MROWS * HH;                      // [M, 768] bf16
    u16*   Wb    = Hb + MROWS * HH;                      // [L, 3072, 768] bf16
    float* gates = (float*)(Wb + (size_t)LL * FOURH * HH); // [M, 3072] fp32

    const dim3 cvtGrid(4096), blk(256);
    const dim3 gemmGrid((unsigned)(MROWS / 32), FOURH / 512);   // (4096, 6)
    const dim3 scanGrid((BB * HH) / 256);                        // 192

    // 1) bf16 conversions: X and both layers' W_ih
    cvt_f32_bf16<<<cvtGrid, blk, 0, stream>>>(x, Xb, MROWS * HH);
    cvt_f32_bf16<<<cvtGrid, blk, 0, stream>>>(W_ih, Wb, (long long)LL * FOURH * HH);

    // 2) Layer 0: gates0 = Xb @ W0^T  ->  scan -> h0 (bf16) + finals
    gemm_bf16_wmma<<<gemmGrid, blk, 0, stream>>>(Xb, Wb, gates, HH, FOURH);
    lstm_scan_layer0<<<scanGrid, blk, 0, stream>>>(
        gates, b_ih, b_hh, parts, Hb, hfin, cfin);

    // 3) Layer 1: gates1 = h0 @ W1^T  ->  scan -> outputs + finals
    gemm_bf16_wmma<<<gemmGrid, blk, 0, stream>>>(
        Hb, Wb + (size_t)FOURH * HH, gates, HH, FOURH);
    lstm_scan_layer1<<<scanGrid, blk, 0, stream>>>(
        gates, b_ih + FOURH, b_hh + FOURH, out, hfin + (size_t)BB * HH,
        cfin + (size_t)BB * HH);
}
